// LinearSeqAttnAction2_11175504904513
// MI455X (gfx1250) — matrix operand order repaired and checked by reading, not compiled, not run on hardware
//
#include <hip/hip_runtime.h>

// Problem constants (from reference): B=64, L=8192, D=256, A=16, H=128
#define B_ 64
#define L_ 8192
#define D_ 256
#define H_ 128

typedef __attribute__((ext_vector_type(16))) __bf16 v16bf;
typedef __attribute__((ext_vector_type(8)))  __bf16 v8bf;
typedef __attribute__((ext_vector_type(4)))  __bf16 v4bf;
typedef __attribute__((ext_vector_type(8)))  float  v8f;

#define NTHREADS 128          // 4 waves
#define ROWS   128            // rows (M) per chunk (4 waves x 2 M-tiles x 16)
#define CHUNKS 4              // chunks per workgroup -> 512 rows per WG
#define KPAD   8              // pad rows by 8 bf16 (16B) for alignment / banks
#define XSTR   (D_ + KPAD)    // 264 bf16 per x row in LDS
#define WSTR   (D_ + KPAD)    // 264 bf16 per w1t row (one output column n)

// LDS partition (bytes):
//  w1t : H_   * WSTR * 2 = 67584
//  xt  : ROWS * XSTR * 2 = 67584
//  b1  : 128*4 = 512
//  w2  : 128*4 = 512          total = 136192  (two WGs fit in 320KB WGP)
#define LDS_BYTES (H_*WSTR*2 + ROWS*XSTR*2 + 2*H_*4)

union ABfrag { v16bf v; v8bf h[2]; };

__global__ __launch_bounds__(NTHREADS) void score_kernel(
    const float* __restrict__ x, const int* __restrict__ actions,
    const float* __restrict__ w1, const float* __restrict__ b1,
    const float* __restrict__ w2, const float* __restrict__ b2,
    float* __restrict__ scores)
{
  extern __shared__ char smem[];
  __bf16* w1t = (__bf16*)smem;                       // [H_][WSTR]  (n-major, k contiguous)
  __bf16* xt  = (__bf16*)(smem + H_*WSTR*2);         // [ROWS][XSTR]
  float*  b1s = (float*)(smem + H_*WSTR*2 + ROWS*XSTR*2);
  float*  w2s = b1s + H_;

  const int b    = blockIdx.y;
  const int act  = actions[b];
  const int tid  = threadIdx.x;
  const int lane = tid & 31;
  const int wave = tid >> 5;          // 0..3
  const int colL = lane & 15;
  const int half = lane >> 4;

  // ---- one-time: load + transpose w1[act] (D_ x H_, row-major) -> bf16 [n][k] ----
  const float*  w1a = w1 + (size_t)act * D_ * H_;
  const float4* w14 = (const float4*)w1a;
  for (int i4 = tid; i4 < (D_*H_)/4; i4 += NTHREADS) {
    float4 v = w14[i4];
    int flat = i4 * 4;
    int k = flat >> 7;         // / H_
    int n = flat & (H_ - 1);
    w1t[(n+0)*WSTR + k] = (__bf16)v.x;
    w1t[(n+1)*WSTR + k] = (__bf16)v.y;
    w1t[(n+2)*WSTR + k] = (__bf16)v.z;
    w1t[(n+3)*WSTR + k] = (__bf16)v.w;
  }
  if (tid < H_) {
    b1s[tid] = b1[act*H_ + tid];
    w2s[tid] = w2[act*H_ + tid];   // w2 is (A,H,1) -> flat A*H
  }
  __syncthreads();

  // per-lane epilogue constants: this lane's column within each 16-wide N tile
  float b1v[8], w2v[8];
  #pragma unroll
  for (int nt = 0; nt < 8; ++nt) {
    b1v[nt] = b1s[nt*16 + colL];
    w2v[nt] = w2s[nt*16 + colL];
  }
  const float b2v = b2[act];

  const int m0    = wave * 32;
  const int arow0 = m0 + colL;           // A tile 0: lane -> matrix row M
  const int arow1 = arow0 + 16;          // A tile 1
  const int kA    = half * 8;            // A: lanes 16-31 hold K 8..15 / 24..31

  for (int c = 0; c < CHUNKS; ++c) {
    const int r0 = (blockIdx.x * CHUNKS + c) * ROWS;      // row base within L
    const float*  xrow = x + ((size_t)b * L_ + r0) * D_;
    const float4* x4   = (const float4*)xrow;

    // ---- cooperative, coalesced stage of 128x256 fp32 x-rows -> bf16 LDS ----
    for (int i4 = tid; i4 < (ROWS*D_)/4; i4 += NTHREADS) {
      float4 v = x4[i4];
      int flat = i4 * 4;
      int row  = flat >> 8;        // / D_
      int k    = flat & (D_ - 1);
      v4bf bv;
      bv[0] = (__bf16)v.x; bv[1] = (__bf16)v.y;
      bv[2] = (__bf16)v.z; bv[3] = (__bf16)v.w;
      *(v4bf*)&xt[row*XSTR + k] = bv;
    }
    if (c + 1 < CHUNKS) __builtin_prefetch(xrow + (size_t)ROWS * D_, 0, 1);
    __syncthreads();

    // ---- WMMA main loop: 64 steps (8 kt x 8 nt), two M-tiles per B fragment.
    //      Software-pipelined: loads for step s+1 are issued into the *other*
    //      buffer before step s's WMMAs, so no zero-distance s_wait_dscnt.
    v8f acc0[8], acc1[8];
    #pragma unroll
    for (int nt = 0; nt < 8; ++nt) { acc0[nt] = 0.0f; acc1[nt] = 0.0f; }

    ABfrag a0buf[2], a1buf[2], bbuf[2];

    // prologue: A for kt=0 into buffer 0, B for (0,0) into buffer 0
    {
      const int kb = 0*32 + kA;
      a0buf[0].h[0] = *(const v8bf*)&xt[arow0*XSTR + kb];
      a0buf[0].h[1] = *(const v8bf*)&xt[arow0*XSTR + kb + 16];
      a1buf[0].h[0] = *(const v8bf*)&xt[arow1*XSTR + kb];
      a1buf[0].h[1] = *(const v8bf*)&xt[arow1*XSTR + kb + 16];
      const int n0  = colL;
      const int kbB = half*8;
      bbuf[0].h[0] = *(const v8bf*)&w1t[n0*WSTR + kbB];
      bbuf[0].h[1] = *(const v8bf*)&w1t[n0*WSTR + kbB + 16];
    }

    #pragma unroll
    for (int s = 0; s < 64; ++s) {
      const int kt  = s >> 3;
      const int nt  = s & 7;
      const int cur = s & 1;
      const int nxt = cur ^ 1;

      if (s + 1 < 64) {                  // issue next step's loads first
        const int nkt = (s + 1) >> 3;
        const int nnt = (s + 1) & 7;
        if (nnt == 0) {                  // new K step: refill A ping-pong
          const int kb = nkt*32 + kA;
          const int ab = nkt & 1;
          a0buf[ab].h[0] = *(const v8bf*)&xt[arow0*XSTR + kb];
          a0buf[ab].h[1] = *(const v8bf*)&xt[arow0*XSTR + kb + 16];
          a1buf[ab].h[0] = *(const v8bf*)&xt[arow1*XSTR + kb];
          a1buf[ab].h[1] = *(const v8bf*)&xt[arow1*XSTR + kb + 16];
        }
        const int n   = nnt*16 + colL;
        const int kbB = nkt*32 + half*8;
        bbuf[nxt].h[0] = *(const v8bf*)&w1t[n*WSTR + kbB];
        bbuf[nxt].h[1] = *(const v8bf*)&w1t[n*WSTR + kbB + 16];
      }

      const int ab = kt & 1;
      acc0[nt] = __builtin_amdgcn_wmma_f32_16x16x32_bf16(
          false, a0buf[ab].v, false, bbuf[cur].v, (short)0, acc0[nt], false, false);
      acc1[nt] = __builtin_amdgcn_wmma_f32_16x16x32_bf16(
          false, a1buf[ab].v, false, bbuf[cur].v, (short)0, acc1[nt], false, false);
    }

    // ---- epilogue: relu(h + b1) . w2, reduce 16 lanes -> one score per row ----
    // C layout: VGPR r, lanes 0-15: (M=r, N=lane); lanes 16-31: (M=r+8, N=lane-16)
    #pragma unroll
    for (int t = 0; t < 2; ++t) {
      #pragma unroll
      for (int r = 0; r < 8; ++r) {
        float s = 0.0f;
        #pragma unroll
        for (int nt = 0; nt < 8; ++nt) {
          float hv = (t ? acc1[nt][r] : acc0[nt][r]) + b1v[nt];
          s += fmaxf(hv, 0.0f) * w2v[nt];
        }
        #pragma unroll
        for (int off = 8; off > 0; off >>= 1)
          s += __shfl_xor(s, off, 16);     // reduce within each 16-lane half
        if (colL == 0) {
          int row = r0 + m0 + t*16 + r + half*8;
          scores[(size_t)b * L_ + row] = s + b2v;
        }
      }
    }
    __syncthreads();   // xt reused next chunk
  }
}

__global__ __launch_bounds__(256) void softmax_kernel(
    const float* __restrict__ scores, const unsigned char* __restrict__ mask,
    float* __restrict__ out)
{
  __shared__ float red[256];
  const int b   = blockIdx.x;
  const int tid = threadIdx.x;
  const float* s = scores + (size_t)b * L_;
  const unsigned char* m = mask + (size_t)b * L_;

  float mx = -INFINITY;
  for (int i = tid; i < L_; i += 256) {
    float v = m[i] ? -INFINITY : s[i];
    mx = fmaxf(mx, v);
  }
  red[tid] = mx; __syncthreads();
  for (int st = 128; st > 0; st >>= 1) {
    if (tid < st) red[tid] = fmaxf(red[tid], red[tid + st]);
    __syncthreads();
  }
  mx = red[0]; __syncthreads();

  float sum = 0.0f;
  for (int i = tid; i < L_; i += 256) {
    float v = m[i] ? -INFINITY : s[i];
    sum += __expf(v - mx);
  }
  red[tid] = sum; __syncthreads();
  for (int st = 128; st > 0; st >>= 1) {
    if (tid < st) red[tid] += red[tid + st];
    __syncthreads();
  }
  const float inv = 1.0f / red[0];

  for (int i = tid; i < L_; i += 256) {
    float v = m[i] ? -INFINITY : s[i];
    out[(size_t)b * L_ + i] = __expf(v - mx) * inv;
  }
}

extern "C" void kernel_launch(void* const* d_in, const int* in_sizes, int n_in,
                              void* d_out, int out_size, void* d_ws, size_t ws_size,
                              hipStream_t stream) {
  (void)in_sizes; (void)n_in; (void)out_size; (void)ws_size;
  const float*         x      = (const float*)d_in[0];
  const unsigned char* x_mask = (const unsigned char*)d_in[1];
  const int*           act    = (const int*)d_in[2];
  const float*         w1     = (const float*)d_in[3];
  const float*         b1     = (const float*)d_in[4];
  const float*         w2     = (const float*)d_in[5];
  const float*         b2     = (const float*)d_in[6];
  float* scores = (float*)d_ws;        // 64*8192 floats = 2 MB scratch
  float* out    = (float*)d_out;

  // allow >64KB dynamic LDS (320KB per WGP on CDNA5)
  (void)hipFuncSetAttribute((const void*)score_kernel,
                            hipFuncAttributeMaxDynamicSharedMemorySize, LDS_BYTES);

  dim3 grid(L_ / (ROWS * CHUNKS), B_);   // 16 x 64 = 1024 WGs
  score_kernel<<<grid, NTHREADS, LDS_BYTES, stream>>>(x, act, w1, b1, w2, b2, scores);
  softmax_kernel<<<B_, 256, 0, stream>>>(scores, x_mask, out);
}
